// MixformerAttention_81200651698759
// MI455X (gfx1250) — matrix operand order repaired and checked by reading, not compiled, not run on hardware
//
#include <hip/hip_runtime.h>
#include <hip/hip_bf16.h>
#include <stdint.h>

// ---------- CDNA5 WMMA types ----------
typedef __attribute__((ext_vector_type(16))) __bf16 v16bf;
typedef __attribute__((ext_vector_type(8)))  float  v8f;
typedef __attribute__((ext_vector_type(4)))  float  f32x4;
typedef __attribute__((ext_vector_type(4)))  unsigned int u32x4;
typedef __attribute__((ext_vector_type(2)))  unsigned int u32x2;
typedef __attribute__((ext_vector_type(4)))  int i32x4;
typedef unsigned short u16;

#if __has_builtin(__builtin_amdgcn_global_load_async_to_lds_b128) && \
    __has_builtin(__builtin_amdgcn_s_wait_asynccnt)
#define USE_ASYNC_LDS 1
// pointer types the async builtin actually wants (from hipcc diagnostics):
typedef __attribute__((address_space(1))) i32x4* gbl_v4i_p;
typedef __attribute__((address_space(3))) i32x4* lds_v4i_p;
#else
#define USE_ASYNC_LDS 0
#endif

union FragBF {            // 16x32 (A) or 32x16 (B) bf16 fragment = 8 VGPRs
  u32x4 q[2];
  v16bf v;
};

__device__ __forceinline__ u16 f2bf(float f) {   // f32 -> bf16, round-nearest-even
  unsigned u = __builtin_bit_cast(unsigned, f);
  u += 0x7FFFu + ((u >> 16) & 1u);
  return (u16)(u >> 16);
}

// two f32 -> packed bf16 pair (lo = a, hi = b)
__device__ __forceinline__ unsigned pk_bf16(float a, float b) {
#if __has_builtin(__builtin_amdgcn_cvt_pk_bf16_f32)
  typedef __attribute__((ext_vector_type(2))) __bf16 v2bf;
  v2bf r = __builtin_amdgcn_cvt_pk_bf16_f32(a, b);
  return __builtin_bit_cast(unsigned, r);
#else
  return (unsigned)f2bf(a) | ((unsigned)f2bf(b) << 16);
#endif
}

__device__ __forceinline__ v8f wmma_bf16(const FragBF& a, const FragBF& b, v8f c) {
  // D = A(16x32 bf16) * B(32x16 bf16) + C(16x16 f32)
  return __builtin_amdgcn_wmma_f32_16x16x32_bf16(false, a.v, false, b.v,
                                                 (short)0, c, false, false);
}

// =====================================================================
// Tiled GEMM: Out(MxN) = A(MxK) * B(KxN) [+ bias], A f32 or bf16, B f32.
// Block = 256 threads (8 wave32), tile 128x128, K-step 32.
// Each wave owns one 16-wide column strip across all 8 row tiles.
// =====================================================================
template<int KDIM, bool A_F32, bool OUT_BF16, bool BIAS>
__global__ __launch_bounds__(256)
void gemm_wmma_kernel(const void* __restrict__ Aptr,
                      const float* __restrict__ Bw,
                      const float* __restrict__ bias,
                      void* __restrict__ Out, int Nout)
{
  // Row stride 40 halfwords keeps every 8-halfword fragment chunk 16B aligned.
  __shared__ u16 As[128][40];
  __shared__ u16 Bs[128][40];   // stored transposed: Bs[n][k]

  const int tid  = threadIdx.x;
  const int wave = tid >> 5;
  const int lane = tid & 31;
  const int half = lane >> 4;   // lane group (0: lanes 0-15, 1: lanes 16-31)
  const int l16  = lane & 15;
  const int m0 = blockIdx.y * 128;
  const int n0 = blockIdx.x * 128;

  v8f acc[8];
  #pragma unroll
  for (int i = 0; i < 8; ++i) { v8f z = {}; acc[i] = z; }

  for (int kk = 0; kk < KDIM; kk += 32) {
    // ---- stage A tile (128x32) into LDS as bf16 ----
    if constexpr (A_F32) {
      #pragma unroll
      for (int j = 0; j < 4; ++j) {                 // 1024 float4 chunks
        int idx = tid + 256 * j;
        int r = idx >> 3, c4 = (idx & 7) * 4;
        f32x4 v = *(const f32x4*)((const float*)Aptr +
                                  (size_t)(m0 + r) * KDIM + kk + c4);
        u32x2 p = { pk_bf16(v.x, v.y), pk_bf16(v.z, v.w) };
        *(u32x2*)&As[r][c4] = p;                    // b64 LDS store
      }
    } else {
      // A already bf16: pure copy -> gfx1250 async global->LDS when available
      #pragma unroll
      for (int j = 0; j < 2; ++j) {                 // 512 b128 chunks
        int idx = tid + 256 * j;
        int r = idx >> 2, c8 = (idx & 3) * 8;
        const u16* src = (const u16*)Aptr + (size_t)(m0 + r) * KDIM + kk + c8;
#if USE_ASYNC_LDS
        __builtin_amdgcn_global_load_async_to_lds_b128(
            (gbl_v4i_p)src, (lds_v4i_p)&As[r][c8], 0, 0);
#else
        *(u32x4*)&As[r][c8] = *(const u32x4*)src;
#endif
      }
    }
    // ---- stage B tile (32x128) transposed into LDS as bf16 ----
    #pragma unroll
    for (int j = 0; j < 4; ++j) {                   // 1024 float4 chunks
      int idx = tid + 256 * j;
      int k = idx >> 5, n4 = (idx & 31) * 4;
      f32x4 v = *(const f32x4*)(Bw + (size_t)(kk + k) * Nout + n0 + n4);
      unsigned p0 = pk_bf16(v.x, v.y);
      unsigned p1 = pk_bf16(v.z, v.w);
      Bs[n4 + 0][k] = (u16)p0;
      Bs[n4 + 1][k] = (u16)(p0 >> 16);              // ds_store_b16_d16_hi
      Bs[n4 + 2][k] = (u16)p1;
      Bs[n4 + 3][k] = (u16)(p1 >> 16);
    }
    if (kk + 32 < KDIM) {  // gfx1250 global_prefetch of next K slab
      __builtin_prefetch((const char*)Aptr +
          ((size_t)(m0 + (tid >> 1)) * KDIM + kk + 32) * (A_F32 ? 4 : 2), 0, 3);
      __builtin_prefetch((const char*)Bw +
          ((size_t)(kk + 32 + (tid >> 6)) * Nout + n0 + (tid & 63) * 2) * 4, 0, 3);
    }
#if USE_ASYNC_LDS
    if constexpr (!A_F32) __builtin_amdgcn_s_wait_asynccnt(0);
#endif
    __syncthreads();

    // ---- wave compute: 8 WMMA per K-step ----
    FragBF bf;   // B fragment: lane -> N=l16, elements -> K in {8h..8h+7, 16+8h..}
    const u32x4* bp = (const u32x4*)&Bs[wave * 16 + l16][8 * half];
    bf.q[0] = bp[0];
    bf.q[1] = bp[2];
    #pragma unroll
    for (int mt = 0; mt < 8; ++mt) {
      FragBF af; // A fragment: lane -> M=l16, elements -> K in {8h..8h+7, 16+8h..}
      af.q[0] = *(const u32x4*)&As[mt * 16 + l16][8 * half];
      af.q[1] = *(const u32x4*)&As[mt * 16 + l16][16 + 8 * half];
      acc[mt] = wmma_bf16(af, bf, acc[mt]);
    }
    __syncthreads();
  }

  // ---- store: C/D layout row = r + 8*half, col = l16 ----
  const int col = n0 + wave * 16 + l16;
  const float bv = BIAS ? bias[col] : 0.0f;
  #pragma unroll
  for (int mt = 0; mt < 8; ++mt) {
    #pragma unroll
    for (int r = 0; r < 8; ++r) {
      int row = m0 + mt * 16 + r + 8 * half;
      float val = acc[mt][r] + bv;
      if (OUT_BF16) ((u16*)Out)[(size_t)row * Nout + col] = f2bf(val);
      else          ((float*)Out)[(size_t)row * Nout + col] = val;
    }
  }
}

// =====================================================================
// Mixformer attention. qkv: (B=64, N=320, 3, H=12, Dh=64) bf16 (= GEMM1
// row-major output). One block per (b, h, qtile) with qtile in [0,5):
//   qtile 0 -> queries 0..63 over keys 0..63   (window / "mt" part)
//   qtile 1..4 -> 64 queries over all 320 keys (global / "s" part)
// =====================================================================
__global__ __launch_bounds__(256)
void attn_kernel(const u16* __restrict__ qkv, u16* __restrict__ attnOut)
{
  __shared__ float S[64][320];   // f32 scores / exp values
  __shared__ u16   P[64][320];   // softmax probs, bf16
  __shared__ u16   VT[64][320];  // V transposed: VT[d][token], bf16

  const int blk = blockIdx.x;           // b*60 + h*5 + qt
  const int b   = blk / 60;
  const int rem = blk % 60;
  const int h   = rem / 5;
  const int qt  = rem % 5;
  const int kvLen = (qt == 0) ? 64 : 320;
  const int q0    = qt * 64;

  const int tid  = threadIdx.x;
  const int wave = tid >> 5, lane = tid & 31;
  const int half = lane >> 4, l16 = lane & 15;

  const u16* Qb = qkv + ((size_t)b * 320 + q0) * 2304 + h * 64;          // q
  const u16* Kb = qkv + (size_t)b * 320 * 2304 + 768  + h * 64;          // k
  const u16* Vb = qkv + (size_t)b * 320 * 2304 + 1536 + h * 64;          // v

  // ---- stage V^T into LDS ----
  for (int lin = tid; lin < kvLen * 16; lin += 256) {
    int t = lin >> 4;
    int d = (lin & 15) * 4;
    unsigned long long w = *(const unsigned long long*)(Vb + (size_t)t * 2304 + d);
    VT[d + 0][t] = (u16)(w);
    VT[d + 1][t] = (u16)(w >> 16);
    VT[d + 2][t] = (u16)(w >> 32);
    VT[d + 3][t] = (u16)(w >> 48);
  }

  // ---- S = scale * Q @ K^T : wave owns one q-tile, Q fragments hoisted ----
  {
    const int nKT = kvLen >> 4;
    const int qT = wave & 3;                 // 2 waves per q-tile
    FragBF aQ[2];
    #pragma unroll
    for (int s = 0; s < 2; ++s) {            // Dh=64 -> 2 K-steps of 32
      const u16* qp = Qb + (size_t)(qT * 16 + l16) * 2304 + s * 32 + 8 * half;
      aQ[s].q[0] = *(const u32x4*)qp;
      aQ[s].q[1] = *(const u32x4*)(qp + 16);
    }
    for (int kT = wave >> 2; kT < nKT; kT += 2) {
      v8f acc = {};
      #pragma unroll
      for (int s = 0; s < 2; ++s) {
        FragBF bf;
        const u16* kp = Kb + (size_t)(kT * 16 + l16) * 2304 + s * 32 + 8 * half;
        bf.q[0] = *(const u32x4*)kp;
        bf.q[1] = *(const u32x4*)(kp + 16);
        acc = wmma_bf16(aQ[s], bf, acc);
      }
      #pragma unroll
      for (int r = 0; r < 8; ++r)
        S[qT * 16 + r + 8 * half][kT * 16 + l16] = acc[r] * 0.125f;  // Dh^-0.5
    }
  }
  __syncthreads();

  // ---- softmax: 4 lanes per row, shfl_xor reductions ----
  {
    const int row = tid >> 2, part = tid & 3;
    const int chunk = kvLen >> 2, c0 = part * chunk;
    float mx = -3.0e38f;
    for (int k = c0; k < c0 + chunk; ++k) mx = fmaxf(mx, S[row][k]);
    mx = fmaxf(mx, __shfl_xor(mx, 1, 4));
    mx = fmaxf(mx, __shfl_xor(mx, 2, 4));
    float sum = 0.0f;
    for (int k = c0; k < c0 + chunk; ++k) {
      float e = __expf(S[row][k] - mx);
      S[row][k] = e;
      sum += e;
    }
    sum += __shfl_xor(sum, 1, 4);
    sum += __shfl_xor(sum, 2, 4);
    float inv = 1.0f / sum;
    for (int k = c0; k < c0 + chunk; k += 2)     // packed bf16 pair stores
      *(unsigned*)&P[row][k] = pk_bf16(S[row][k] * inv, S[row][k + 1] * inv);
  }
  __syncthreads();

  // ---- O = P @ V  (16 output tiles, 2 per wave) ----
  for (int t = wave; t < 16; t += 8) {
    int qT = t >> 2, dT = t & 3;
    v8f acc = {};
    for (int k0 = 0; k0 < kvLen; k0 += 32) {
      FragBF af, bf;
      const u16* pp = &P[qT * 16 + l16][k0 + 8 * half];
      af.q[0] = *(const u32x4*)pp;
      af.q[1] = *(const u32x4*)(pp + 16);
      const u16* vp = &VT[dT * 16 + l16][k0 + 8 * half];
      bf.q[0] = *(const u32x4*)vp;
      bf.q[1] = *(const u32x4*)(vp + 16);
      acc = wmma_bf16(af, bf, acc);
    }
    #pragma unroll
    for (int r = 0; r < 8; ++r) {
      int tok = q0 + qT * 16 + r + 8 * half;
      int col = h * 64 + dT * 16 + l16;     // (B, token, H*Dh) layout
      attnOut[((size_t)b * 320 + tok) * 768 + col] = f2bf(acc[r]);
    }
  }
}

// =====================================================================
extern "C" void kernel_launch(void* const* d_in, const int* in_sizes, int n_in,
                              void* d_out, int out_size, void* d_ws, size_t ws_size,
                              hipStream_t stream) {
  (void)in_sizes; (void)n_in; (void)out_size; (void)ws_size;
  const float* x     = (const float*)d_in[0];   // (64,320,768) f32
  const float* Wqkv  = (const float*)d_in[1];   // (768,2304)   f32
  const float* Wproj = (const float*)d_in[2];   // (768,768)    f32
  const float* bproj = (const float*)d_in[3];   // (768,)       f32
  float* out = (float*)d_out;                   // (64,320,768) f32

  const size_t M = 64 * 320;                    // 20480 rows
  u16* qkv  = (u16*)d_ws;                       // M x 2304 bf16 (94.4 MB)
  u16* attn = qkv + M * 2304;                   // M x 768  bf16 (31.5 MB)

  // 1) qkv = bf16(x @ W_qkv)
  gemm_wmma_kernel<768, true, true, false>
      <<<dim3(2304 / 128, (unsigned)(M / 128)), 256, 0, stream>>>(
          x, Wqkv, nullptr, qkv, 2304);

  // 2) mixed window/global attention -> attn (bf16, head-interleaved)
  attn_kernel<<<64 * 12 * 5, 256, 0, stream>>>(qkv, attn);

  // 3) out = attn @ W_proj + b_proj (f32 output)
  gemm_wmma_kernel<768, false, false, true>
      <<<dim3(768 / 128, (unsigned)(M / 128)), 256, 0, stream>>>(
          attn, Wproj, bproj, out, 768);
}
